// decoder_1005022347819
// MI455X (gfx1250) — matrix operand order repaired and checked by reading, not compiled, not run on hardware
//
#include <hip/hip_runtime.h>

typedef __attribute__((ext_vector_type(16))) _Float16 v16h;
typedef __attribute__((ext_vector_type(8)))  _Float16 h8;
typedef __attribute__((ext_vector_type(4)))  _Float16 h4;
typedef __attribute__((ext_vector_type(8)))  float    v8f;

#define QCH   16    // channels
#define LLEN  512   // sequence length
#define KTAPS 65    // filter taps
#define TILE  256   // outputs per workgroup (16x16 WMMA D tile)
#define NKC   3     // K chunks of 32 (Toeplitz K' = 80 padded to 96)
#define XSPAN 336   // x halo per channel (16B-aligned rows: 336*2 = 672 B)
#define PADL  32
#define NFRAG (QCH * NKC * 32)   // 1536 B-fragments

// ---------------------------------------------------------------------------
// Builder: (1) bake banded-Toeplitz B[k,n] = wflip[k-n] (wflip[j]=w[64-j])
// into per-lane WMMA B-fragment order (f16, 48 KB, L2-resident afterwards);
// (2) bake the sequential at-chain at[q] = A[q]/at[q-1] and sqrt(at[q])
// so the hot scan needs no divides for it.
// ---------------------------------------------------------------------------
__global__ __launch_bounds__(256)
void build_tables_kernel(const float* __restrict__ weight,
                         const float* __restrict__ Avec,
                         _Float16* __restrict__ bt,
                         float* __restrict__ scal)
{
    const int idx  = blockIdx.x * 256 + threadIdx.x;   // 0..1535
    const int lane = idx & 31;
    const int kc   = (idx >> 5) % NKC;
    const int q    = idx / (32 * NKC);
    const int lo   = lane & 15;      // N
    const int hi   = lane >> 4;      // K half-select
    const int k0   = kc * 32;

    v16h frag;
    #pragma unroll
    for (int h = 0; h < 16; ++h) {
        int k = k0 + (h & 7) + 8 * hi + 16 * (h >> 3);
        int j = k - lo;
        float v = (j >= 0 && j < KTAPS) ? weight[q * KTAPS + (KTAPS - 1 - j)] : 0.f;
        frag[h] = (_Float16)v;
    }
    *(v16h*)(bt + (size_t)idx * 16) = frag;

    if (idx == 0) {                      // exact fp32, once
        float at = 1.f;
        for (int qq = 0; qq < QCH; ++qq) {
            at = Avec[qq] / at;
            scal[2 * qq]     = at;
            scal[2 * qq + 1] = sqrtf(at);
        }
    }
}

// ---------------------------------------------------------------------------
// Fused: depthwise conv via Toeplitz WMMA (f16 in, f32 acc) + 16-step scan.
// ---------------------------------------------------------------------------
__global__ __launch_bounds__(256)
void fused_dwconv_scan_kernel(const float* __restrict__ x,
                              const _Float16* __restrict__ bt,
                              const float* __restrict__ scal,
                              float* __restrict__ out)
{
    __shared__ alignas(16) _Float16 xs[QCH][XSPAN];  // 10.5 KB
    __shared__ alignas(32) float    yt[QCH * TILE];  // 16.0 KB conv outputs

    const int tid = threadIdx.x;
    const int b   = blockIdx.x >> 1;
    const int l0  = (blockIdx.x & 1) * TILE;

    // ---- stage x halo -> LDS f16: 16 threads per channel row, branchless ----
    {
        const int sq = tid >> 4;            // staging channel
        const int sc = tid & 15;            // chunk slot in row
        const float* xrowg = x + (long)b * (QCH * LLEN) + (long)sq * LLEN;
        #pragma unroll
        for (int it = 0; it < 6; ++it) {
            int p4i = sc + it * 16;
            if (it < 5 || p4i < XSPAN / 4) {     // only last iter partial
                int p4   = p4i * 4;
                int gpos = l0 - PADL + p4;       // chunk fully in or fully out
                int gc   = gpos < 0 ? 0 : (gpos > LLEN - 4 ? LLEN - 4 : gpos);
                float4 v = *(const float4*)(xrowg + gc);     // global_load_b128
                bool ok  = (gpos >= 0) & (gpos < LLEN);
                h4 hv = { (_Float16)(ok ? v.x : 0.f), (_Float16)(ok ? v.y : 0.f),
                          (_Float16)(ok ? v.z : 0.f), (_Float16)(ok ? v.w : 0.f) };
                *(h4*)&xs[sq][p4] = hv;                      // ds_store_b64
            }
        }
    }
    __syncthreads();

    const int wave = tid >> 5;
    const int lane = tid & 31;
    const int lo   = lane & 15;   // A: M, B/C/D: N
    const int hi   = lane >> 4;

    // ---- WMMA: 8 waves x 2 channels x 3 chunks of K=32 ----
    const v16h* __restrict__ btv = (const v16h*)bt;
    #pragma unroll
    for (int qq = 0; qq < 2; ++qq) {
        const int q = wave * 2 + qq;
        const h8* __restrict__ xrow = (const h8*)&xs[q][0];
        v8f acc = {};
        #pragma unroll
        for (int kc = 0; kc < NKC; ++kc) {
            const int k0 = kc * 32;
            h8 a0 = xrow[lo * 2 + (k0 >> 3) + hi];           // ds_load_b128
            h8 a1 = xrow[lo * 2 + (k0 >> 3) + 2 + hi];       // ds_load_b128
            v16h a = __builtin_shufflevector(a0, a1,
                     0, 1, 2, 3, 4, 5, 6, 7, 8, 9, 10, 11, 12, 13, 14, 15);
            v16h bm = btv[(q * NKC + kc) * 32 + lane];       // 2x global_load_b128
            acc = __builtin_amdgcn_wmma_f32_16x16x32_f16(
                false, a, false, bm, (short)0, acc, false, false);
        }
        // contiguous 32B per lane; scan un-permutes via ridx below
        *(v8f*)&yt[q * TILE + lane * 8] = acc;               // 2x ds_store_b128
    }
    __syncthreads();

    // ---- 16-step recurrence; thread tid owns position l0+tid ----
    // stored at [lane*8 + v] with lane = 16*(m>>3)+n, v = m&7, tid = 16m+n
    const int ridx = ((((tid >> 7) << 4) | (tid & 15)) << 3) | ((tid >> 4) & 7);
    float r = 0.f, t = 1.f;
    #pragma unroll
    for (int q = 0; q < QCH; ++q) {
        float atq  = scal[2 * q];          // uniform -> s_load
        float satq = scal[2 * q + 1];
        float rn   = yt[q * TILE + ridx];
        float fab  = __builtin_amdgcn_rcpf(1.f - atq * r * rn);
        float rnw  = r + atq * rn * t * t * fab;
        t = satq * t * (1.f - rn) * fab;
        r = rnw;
    }
    out[(long)b * LLEN + l0 + tid] = r;
}

extern "C" void kernel_launch(void* const* d_in, const int* in_sizes, int n_in,
                              void* d_out, int out_size, void* d_ws, size_t ws_size,
                              hipStream_t stream) {
    const float* x = (const float*)d_in[0];   // (B, 16, 512) fp32
    const float* w = (const float*)d_in[1];   // (16, 1, 65)  fp32
    const float* A = (const float*)d_in[2];   // (16,)        fp32
    float* out = (float*)d_out;               // (B, 512)     fp32

    _Float16* bt  = (_Float16*)d_ws;                  // 48 KB B-fragments
    float*    scal = (float*)(bt + (size_t)NFRAG * 16); // 32 floats: at, sqrt(at)

    build_tables_kernel<<<dim3(NFRAG / 256), dim3(256), 0, stream>>>(w, A, bt, scal);

    const int batch  = in_sizes[0] / (QCH * LLEN);
    const int blocks = batch * (LLEN / TILE);
    fused_dwconv_scan_kernel<<<dim3(blocks), dim3(256), 0, stream>>>(x, bt, scal, out);
}
